// VisualFeatureExtractor_386547056806
// MI455X (gfx1250) — matrix operand order repaired and checked by reading, not compiled, not run on hardware
//
#include <hip/hip_runtime.h>
#include <hip/hip_bf16.h>

// CDNA5 / gfx1250, wave32. GEMM via v_wmma_f32_16x16x32_bf16.

typedef __attribute__((ext_vector_type(16))) __bf16 v16bf;
typedef __attribute__((ext_vector_type(8)))  float  v8f;
typedef __attribute__((ext_vector_type(4)))  float  f4;   // native vector type

union Frag {
  v16bf bf;
  unsigned short s[16];
  uint4 u4[2];
};

#define SLICE_W    8
#define IMG_H      32
#define IMG_W      8192
#define BATCH      64
#define SLICES     (BATCH * (IMG_W / SLICE_W))   // 65536
#define KDIM       256                           // H * SLICE_W
#define NDIM       512                           // EMBED_DIM
#define FSTR       264                           // padded bf16 feats row stride (16B aligned, conflict-free)
#define TILES_PER_WG 16

__device__ __forceinline__ unsigned short f2bf(float f) {
  // round-to-nearest-even f32 -> bf16
  unsigned int u = __float_as_uint(f);
  u += 0x7FFFu + ((u >> 16) & 1u);
  return (unsigned short)(u >> 16);
}

__global__ __launch_bounds__(256)
void vfe_conv_bridge_wmma(const float* __restrict__ images,
                          const float* __restrict__ conv_w,
                          const float* __restrict__ conv_b,
                          const float* __restrict__ lin_w,
                          const float* __restrict__ lin_b,
                          float* __restrict__ out) {
  __shared__ float strip[IMG_H][128];            // 16 slices * 32x8 f32 strip (16 KB)
  __shared__ unsigned short feats[16 * FSTR];    // conv output, bf16, padded rows (8.25 KB)

  const int t     = threadIdx.x;
  const int lane  = t & 31;
  const int wave  = t >> 5;
  const int lm    = lane & 15;                   // row (A/M) or col (B/N) within tile
  const int ahalf = (lane >> 4) << 3;            // A: lanes16-31 hold K+8 interleave
  const int bhalf = (lane >> 4) << 4;            // unused; B uses ahalf (same K+8 interleave as A)

  // 3x3 conv weights + bias (uniform -> scalar loads)
  float cw[9];
  #pragma unroll
  for (int i = 0; i < 9; ++i) cw[i] = conv_w[i];
  const float cb = conv_b[0];

  // ---- B fragments: this wave's 16 output columns, full K=256, kept in VGPRs ----
  // lin_w is 512 KB and shared by all workgroups -> default (RT) policy, stays L2-resident.
  const int ncol = blockIdx.y * 128 + wave * 16 + lm;
  Frag bfr[8];
  #pragma unroll
  for (int ks = 0; ks < 8; ++ks) {
    #pragma unroll
    for (int e = 0; e < 16; ++e) {
      const int k = ks * 32 + ahalf + (e & 7) + ((e >> 3) << 4);   // B 32x16 bf16: same K interleave as A (K+8h, K+16+8h)
      bfr[ks].s[e] = f2bf(lin_w[k * NDIM + ncol]);
    }
  }
  const float bias = lin_b[ncol];

  for (int it = 0; it < TILES_PER_WG; ++it) {
    const int tile_m = blockIdx.x * TILES_PER_WG + it;
    const int s0  = tile_m << 4;                 // first of 16 slices (never crosses batch: 16 | 1024)
    const int b   = s0 >> 10;
    const int ws0 = s0 & 1023;
    const size_t gbase = (size_t)b * (IMG_H * IMG_W) + (size_t)ws0 * SLICE_W;

    __syncthreads();                             // prev iter done with strip + feats

    // ---- coalesced load of 32x128 f32 strip into LDS ----
    // RT policy: each strip is re-read by the other 3 N-block workgroups; whole input
    // (64 MB) is L2-resident (192 MB), so re-reads hit L2 instead of HBM.
    #pragma unroll
    for (int j = 0; j < 4; ++j) {
      const int idx4 = t + j * 256;              // 0..1023 float4s
      const int row  = idx4 >> 5;
      const int col  = (idx4 & 31) << 2;
      const f4 v = *(const f4*)&images[gbase + (size_t)row * IMG_W + col];
      *(f4*)&strip[row][col] = v;
    }
    __syncthreads();

    // ---- per-slice 3x3 conv (zero pad inside each 32x8 slice) + bias + ReLU -> bf16 feats ----
    {
      const int m  = t >> 4;                     // slice 0..15
      const int hp = t & 15;                     // row pair
      #pragma unroll
      for (int r = 0; r < 2; ++r) {
        const int hh = hp * 2 + r;
        #pragma unroll
        for (int x = 0; x < SLICE_W; ++x) {
          float acc = cb;
          #pragma unroll
          for (int dh = -1; dh <= 1; ++dh) {
            const int h2 = hh + dh;
            if (h2 < 0 || h2 >= IMG_H) continue;
            #pragma unroll
            for (int dx = -1; dx <= 1; ++dx) {
              const int x2 = x + dx;
              if (x2 < 0 || x2 >= SLICE_W) continue;   // compile-time after unroll
              acc += cw[(dh + 1) * 3 + (dx + 1)] * strip[h2][m * SLICE_W + x2];
            }
          }
          feats[m * FSTR + hh * SLICE_W + x] = f2bf(acc > 0.f ? acc : 0.f);
        }
      }
    }
    __syncthreads();

    // ---- preload ALL 8 A fragments as one ds clause (single dscnt wait), ----
    // ---- fenced with sched barriers so the scheduler can't re-interleave, ----
    // ---- then chain the 8 WMMAs back-to-back on the matrix pipe.          ----
    __builtin_amdgcn_sched_barrier(0);
    Frag a[8];
    const unsigned short* abase = &feats[lm * FSTR + ahalf];
    #pragma unroll
    for (int ks = 0; ks < 8; ++ks) {
      a[ks].u4[0] = *(const uint4*)(abase + ks * 32);        // K +0..7
      a[ks].u4[1] = *(const uint4*)(abase + ks * 32 + 16);   // K +16..23
    }
    __builtin_amdgcn_sched_barrier(0);

    v8f c;
    #pragma unroll
    for (int r = 0; r < 8; ++r) c[r] = bias;

    #pragma unroll
    for (int ks = 0; ks < 8; ++ks) {
      c = __builtin_amdgcn_wmma_f32_16x16x32_bf16(
              /*neg_a=*/false, a[ks].bf, /*neg_b=*/false, bfr[ks].bf,
              /*c_mod=*/(short)0, c, /*reuse_a=*/false, /*reuse_b=*/false);
    }
    __builtin_amdgcn_sched_barrier(0);

    // ---- store D non-temporally (write-once output, don't pollute L2):       ----
    // ---- element (VGPR r, lane L) = row r + 8*(L>>4), col L&15               ----
    float* op = out + (size_t)(s0 + ((lane >> 4) << 3)) * NDIM + ncol;
    #pragma unroll
    for (int r = 0; r < 8; ++r)
      __builtin_nontemporal_store(c[r], op + (size_t)r * NDIM);
  }
}

extern "C" void kernel_launch(void* const* d_in, const int* in_sizes, int n_in,
                              void* d_out, int out_size, void* d_ws, size_t ws_size,
                              hipStream_t stream) {
  const float* images = (const float*)d_in[0];
  const float* conv_w = (const float*)d_in[1];
  const float* conv_b = (const float*)d_in[2];
  const float* lin_w  = (const float*)d_in[3];
  const float* lin_b  = (const float*)d_in[4];
  float* out = (float*)d_out;

  dim3 grid((SLICES / 16) / TILES_PER_WG, NDIM / 128);  // (256, 4)
  dim3 block(256);
  vfe_conv_bridge_wmma<<<grid, block, 0, stream>>>(images, conv_w, conv_b,
                                                   lin_w, lin_b, out);
}
